// DeepHopfield_43997644981027
// MI455X (gfx1250) — matrix operand order repaired and checked by reading, not compiled, not run on hardware
//
#include <hip/hip_runtime.h>
#include <hip/hip_bf16.h>
#include <cstdint>
#include <cstddef>

// ---------------------------------------------------------------------------
// DeepHopfield on MI455X (gfx1250, wave32).
// Dominant cost: 513x [1024,512]@[512,512] GEMM in the relaxation loop
// (~275 GFLOP) -> v_wmma_f32_16x16x32_bf16, operands resident in L2.
// This revision: 2x2 register blocking (32x32/wave, 2 b128 loads per wmma)
// and a fused energy+sign kernel (2 launches per relaxation step, not 3).
// ---------------------------------------------------------------------------

typedef __attribute__((ext_vector_type(16))) __bf16 v16bf;
typedef __attribute__((ext_vector_type(8)))  float  v8f;

#define BATCHN 1024
#define LABELN 128
#define NTOT   1152      // BATCHN + LABELN
#define LATENT 512
#define FEAT   3136      // 64*7*7

__device__ __forceinline__ __hip_bfloat16 f2bf(float x) { return __float2bfloat16(x); }

// --------------------------- conv1 + relu + maxpool2 -----------------------
// in: [N,1,28,28] (N: 0..1023 image, 1024..1151 label_images)
// out: [N,32,14,14] f32
__global__ void conv1_relu_pool(const float* __restrict__ img,
                                const float* __restrict__ limg,
                                const float* __restrict__ w,   // [32,1,5,5]
                                const float* __restrict__ b,
                                float* __restrict__ out) {
  __shared__ float sIn[784];
  __shared__ float sW[25];
  int n = blockIdx.x >> 5;
  int c = blockIdx.x & 31;
  int tid = threadIdx.x;
  const float* src = (n < BATCHN) ? (img + (size_t)n * 784)
                                  : (limg + (size_t)(n - BATCHN) * 784);
  for (int i = tid; i < 784; i += 256) sIn[i] = src[i];
  if (tid < 25) sW[tid] = w[c * 25 + tid];
  __syncthreads();
  if (tid < 196) {
    int ph = tid / 14, pw = tid % 14;
    float bias = b[c];
    float m = 0.0f;  // relu outputs are >= 0
    for (int dy = 0; dy < 2; ++dy)
      for (int dx = 0; dx < 2; ++dx) {
        int h = ph * 2 + dy, wv = pw * 2 + dx;
        float acc = bias;
#pragma unroll
        for (int kh = 0; kh < 5; ++kh) {
          int ih = h + kh - 2;
          if (ih < 0 || ih >= 28) continue;
#pragma unroll
          for (int kw = 0; kw < 5; ++kw) {
            int iw = wv + kw - 2;
            if (iw < 0 || iw >= 28) continue;
            acc += sIn[ih * 28 + iw] * sW[kh * 5 + kw];
          }
        }
        float r = acc > 0.f ? acc : 0.f;
        m = r > m ? r : m;
      }
    out[(size_t)n * 6272 + c * 196 + tid] = m;
  }
}

// --------------------------- conv2 + relu + maxpool2 -----------------------
// in: [N,32,14,14] f32, out: [N, 64*7*7] bf16 (flattened NCHW, fc1-ready)
__global__ void conv2_relu_pool(const float* __restrict__ in,
                                const float* __restrict__ w,   // [64,32,5,5]
                                const float* __restrict__ b,
                                __hip_bfloat16* __restrict__ out) {
  __shared__ float sIn[6272];   // 25 KB
  int n = blockIdx.x >> 2;
  int g = blockIdx.x & 3;       // group of 16 output channels
  int ocb = g << 4;
  int tid = threadIdx.x;
  const float* src = in + (size_t)n * 6272;
  for (int i = tid; i < 6272; i += 256) sIn[i] = src[i];
  __syncthreads();
  for (int idx = tid; idx < 784; idx += 256) {
    int ol = idx / 49, p = idx % 49;
    int oc = ocb + ol;
    int ph = p / 7, pw = p % 7;
    const float* wp = w + (size_t)oc * 800;
    float bias = b[oc];
    float m = 0.f;
    for (int dy = 0; dy < 2; ++dy)
      for (int dx = 0; dx < 2; ++dx) {
        int h = ph * 2 + dy, wv = pw * 2 + dx;
        float acc = bias;
        for (int ic = 0; ic < 32; ++ic) {
          const float* ip = sIn + ic * 196;
          const float* wpc = wp + ic * 25;
#pragma unroll
          for (int kh = 0; kh < 5; ++kh) {
            int ih = h + kh - 2;
            if (ih < 0 || ih >= 14) continue;
#pragma unroll
            for (int kw = 0; kw < 5; ++kw) {
              int iw = wv + kw - 2;
              if (iw < 0 || iw >= 14) continue;
              acc += ip[ih * 14 + iw] * wpc[kh * 5 + kw];
            }
          }
        }
        float r = acc > 0.f ? acc : 0.f;
        m = r > m ? r : m;
      }
    out[(size_t)n * FEAT + oc * 49 + p] = f2bf(m);
  }
}

// --------------------------- WMMA bf16 GEMM (2x2 blocked) ------------------
// C[M,N] (f32) = A[M,K] (bf16, row-major) x Bt[N,K]^T (bf16, row-major in K)
// One wave per 32x32 output block = 4 accumulators; per K-step 8 b128 loads
// feed 4 v_wmma_f32_16x16x32_bf16 (2 loads per wmma).
// Fragment layout per CDNA5 ISA (wave32):
//   A 16x32: lane<16 row l16 K{0..7,16..23}; lane>=16 K{8..15,24..31}
//   B 32x16: lane col l16, same K split as A (offset ha, ha+16)
//   C/D:     VGPR v <-> row v + 8*(lane/16), col lane%16
__global__ __launch_bounds__(256)
void gemm_bf16_wmma(float* __restrict__ C,
                    const __hip_bfloat16* __restrict__ A,
                    const __hip_bfloat16* __restrict__ Bt,
                    int M, int N, int K) {
  int gwave = (int)((blockIdx.x * 256u + threadIdx.x) >> 5);
  int lane = threadIdx.x & 31;
  int tilesN = N >> 5;                   // 32-wide column blocks
  int tM = gwave / tilesN;
  int tN = gwave - tM * tilesN;
  if (tM >= (M >> 5)) return;            // wave-uniform exit (EXEC all-1 at wmma)
  int half = lane >> 4;
  int l16 = lane & 15;
  const __hip_bfloat16* Arow0 = A + (size_t)((tM << 5) + l16) * K;
  const __hip_bfloat16* Arow1 = Arow0 + (size_t)16 * K;
  const __hip_bfloat16* Brow0 = Bt + (size_t)((tN << 5) + l16) * K;
  const __hip_bfloat16* Brow1 = Brow0 + (size_t)16 * K;
  int ha = half << 3;                    // A frag: K offset 0 / 8 (+16 upper 8)
  int hb = half << 4;                    // K offset 0 / 16; unused (B loads use ha)
  union F { v16bf v; uint4 q[2]; };
  F fa0, fa1, fb0, fb1;
  v8f a00 = {0.f, 0.f, 0.f, 0.f, 0.f, 0.f, 0.f, 0.f};
  v8f a01 = a00, a10 = a00, a11 = a00;
#pragma unroll 2
  for (int k0 = 0; k0 < K; k0 += 32) {
    fa0.q[0] = *reinterpret_cast<const uint4*>(Arow0 + k0 + ha);
    fa0.q[1] = *reinterpret_cast<const uint4*>(Arow0 + k0 + ha + 16);
    fa1.q[0] = *reinterpret_cast<const uint4*>(Arow1 + k0 + ha);
    fa1.q[1] = *reinterpret_cast<const uint4*>(Arow1 + k0 + ha + 16);
    fb0.q[0] = *reinterpret_cast<const uint4*>(Brow0 + k0 + ha);
    fb0.q[1] = *reinterpret_cast<const uint4*>(Brow0 + k0 + ha + 16);
    fb1.q[0] = *reinterpret_cast<const uint4*>(Brow1 + k0 + ha);
    fb1.q[1] = *reinterpret_cast<const uint4*>(Brow1 + k0 + ha + 16);
    a00 = __builtin_amdgcn_wmma_f32_16x16x32_bf16(false, fa0.v, false, fb0.v,
                                                  (short)0, a00, false, false);
    a01 = __builtin_amdgcn_wmma_f32_16x16x32_bf16(false, fa0.v, false, fb1.v,
                                                  (short)0, a01, false, false);
    a10 = __builtin_amdgcn_wmma_f32_16x16x32_bf16(false, fa1.v, false, fb0.v,
                                                  (short)0, a10, false, false);
    a11 = __builtin_amdgcn_wmma_f32_16x16x32_bf16(false, fa1.v, false, fb1.v,
                                                  (short)0, a11, false, false);
  }
  int r0 = (tM << 5) + (half << 3);
  int c0 = (tN << 5) + l16;
#pragma unroll
  for (int v = 0; v < 8; ++v) {
    float* Cr = C + (size_t)(r0 + v) * N + c0;
    Cr[0] = a00[v];
    Cr[16] = a01[v];
    float* Cr2 = Cr + (size_t)16 * N;
    Cr2[0] = a10[v];
    Cr2[16] = a11[v];
  }
}

// --------------------------- fc1 epilogue ----------------------------------
// rows [0,1024): lat -> lat_bf, s0 = tanh(lat) (f32 + bf16)
// rows [1024,1152): rep = tanh(.) (f32 + bf16)
__global__ void fc1_epilogue(const float* __restrict__ fc1o, const float* __restrict__ fc1b,
                             __hip_bfloat16* __restrict__ lat_bf,
                             float* __restrict__ s0, __hip_bfloat16* __restrict__ s0_bf,
                             float* __restrict__ rep, __hip_bfloat16* __restrict__ rep_bf) {
  int idx = blockIdx.x * 256 + threadIdx.x;
  if (idx >= NTOT * LATENT) return;
  int col = idx & (LATENT - 1);
  float v = fc1o[idx] + fc1b[col];
  if (idx < BATCHN * LATENT) {
    lat_bf[idx] = f2bf(v);
    float t = tanhf(v);
    s0[idx] = t;
    s0_bf[idx] = f2bf(t);
  } else {
    int r = idx - BATCHN * LATENT;
    float t = tanhf(v);
    rep[r] = t;
    rep_bf[r] = f2bf(t);
  }
}

// --------------------------- reductions for rho ----------------------------
__global__ void reduce_part(const float* __restrict__ x, int n, float* __restrict__ part) {
  __shared__ float sm[256];
  int tid = threadIdx.x;
  float a = 0.f;
  for (int i = blockIdx.x * 256 + tid; i < n; i += gridDim.x * 256) a += x[i];
  sm[tid] = a; __syncthreads();
  for (int st = 128; st > 0; st >>= 1) { if (tid < st) sm[tid] += sm[tid + st]; __syncthreads(); }
  if (tid == 0) part[blockIdx.x] = sm[0];
}

__global__ void reduce_final(const float* __restrict__ part, int nb, float scale,
                             float* __restrict__ o) {
  __shared__ float sm[64];
  int tid = threadIdx.x;
  sm[tid] = (tid < nb) ? part[tid] : 0.f; __syncthreads();
  for (int st = 32; st > 0; st >>= 1) { if (tid < st) sm[tid] += sm[tid + st]; __syncthreads(); }
  if (tid == 0) o[0] = sm[0] * scale;
}

// --------------------------- Hopfield weight -------------------------------
// w[i,j] = (1/128) * sum_b (rep[b,i]-rho)(rep[b,j]-rho), diag zeroed -> bf16
__global__ void hopfield_w(const float* __restrict__ rep, const float* __restrict__ rho_p,
                           __hip_bfloat16* __restrict__ wbf) {
  int idx = blockIdx.x * 256 + threadIdx.x;   // < 512*512
  int i = idx >> 9, j = idx & 511;
  float rho = rho_p[0];
  float acc = 0.f;
  for (int bb = 0; bb < LABELN; ++bb) {
    float a = rep[bb * LATENT + i] - rho;     // broadcast within block
    float c = rep[bb * LATENT + j] - rho;     // coalesced
    acc += a * c;
  }
  wbf[idx] = f2bf((i == j) ? 0.f : acc * (1.0f / LABELN));
}

// --------------------------- relaxation loop pieces ------------------------
__global__ void init_min(float* __restrict__ min_e, __hip_bfloat16* __restrict__ min_s_bf) {
  int idx = blockIdx.x * 256 + threadIdx.x;
  if (idx < BATCHN) min_e[idx] = __builtin_inff();
  if (idx < BATCHN * LATENT) min_s_bf[idx] = f2bf(0.f);
}

// s_new = |s| * sign(z)  (prologue only; steady state uses energy_sign below)
__global__ void sign_update(const float* __restrict__ s, const float* __restrict__ z,
                            float* __restrict__ sn, __hip_bfloat16* __restrict__ sn_bf) {
  int idx = blockIdx.x * 256 + threadIdx.x;
  if (idx >= BATCHN * LATENT) return;
  float zv = z[idx];
  float sg = (zv > 0.f) ? 1.f : ((zv < 0.f) ? -1.f : 0.f);
  float v = fabsf(s[idx]) * sg;
  sn[idx] = v;
  sn_bf[idx] = f2bf(v);
}

// Fused per-step kernel, given z = s @ w:
//   e[row] = -dot(z[row], s[row]); min-track (snapshot s as bf16)
//   s_next = |s| * sign(z)  (f32 + bf16) for the next GEMM
__global__ void energy_sign(const float* __restrict__ z, const float* __restrict__ s,
                            float* __restrict__ min_e, __hip_bfloat16* __restrict__ min_s_bf,
                            float* __restrict__ s_next, __hip_bfloat16* __restrict__ s_next_bf) {
  __shared__ float sm[128];
  __shared__ int flag;
  int row = blockIdx.x, tid = threadIdx.x;
  const size_t base = (size_t)row * LATENT;
  float zv[4], sv[4];
  float a = 0.f;
#pragma unroll
  for (int i = 0; i < 4; ++i) {
    zv[i] = z[base + tid + i * 128];
    sv[i] = s[base + tid + i * 128];
    a += zv[i] * sv[i];
  }
  sm[tid] = a; __syncthreads();
  for (int st = 64; st > 0; st >>= 1) { if (tid < st) sm[tid] += sm[tid + st]; __syncthreads(); }
  if (tid == 0) {
    float e = -sm[0];
    if (e < min_e[row]) { min_e[row] = e; flag = 1; } else flag = 0;
  }
  __syncthreads();
  int f = flag;
#pragma unroll
  for (int i = 0; i < 4; ++i) {
    size_t o = base + tid + i * 128;
    if (f) min_s_bf[o] = f2bf(sv[i]);
    float sg = (zv[i] > 0.f) ? 1.f : ((zv[i] < 0.f) ? -1.f : 0.f);
    float v = fabsf(sv[i]) * sg;
    s_next[o] = v;
    s_next_bf[o] = f2bf(v);
  }
}

// --------------------------- softmax (128-wide rows) -----------------------
// mode 0: x += bias[col]; mode 1: x = |x|
__global__ void softmax128(const float* __restrict__ in, const float* __restrict__ bias,
                           int mode, float* __restrict__ out) {
  __shared__ float sm[128];
  int row = blockIdx.x, tid = threadIdx.x;
  float x = in[(size_t)row * LABELN + tid];
  if (mode == 0) x += bias[tid]; else x = fabsf(x);
  sm[tid] = x; __syncthreads();
  for (int st = 64; st > 0; st >>= 1) { if (tid < st) sm[tid] = fmaxf(sm[tid], sm[tid + st]); __syncthreads(); }
  float mx = sm[0]; __syncthreads();
  float e = expf(x - mx);
  sm[tid] = e; __syncthreads();
  for (int st = 64; st > 0; st >>= 1) { if (tid < st) sm[tid] += sm[tid + st]; __syncthreads(); }
  out[(size_t)row * LABELN + tid] = e / sm[0];
}

// --------------------------- misc ------------------------------------------
__global__ void cvt_bf16(const float* __restrict__ x, __hip_bfloat16* __restrict__ y, int n) {
  int idx = blockIdx.x * 256 + threadIdx.x;
  if (idx < n) y[idx] = f2bf(x[idx]);
}

// ---------------------------------------------------------------------------
extern "C" void kernel_launch(void* const* d_in, const int* in_sizes, int n_in,
                              void* d_out, int out_size, void* d_ws, size_t ws_size,
                              hipStream_t stream) {
  (void)in_sizes; (void)n_in; (void)out_size; (void)ws_size;
  const float* img  = (const float*)d_in[0];
  const float* limg = (const float*)d_in[1];
  const float* c1w  = (const float*)d_in[2];
  const float* c1b  = (const float*)d_in[3];
  const float* c2w  = (const float*)d_in[4];
  const float* c2b  = (const float*)d_in[5];
  const float* fc1w = (const float*)d_in[6];
  const float* fc1b = (const float*)d_in[7];
  const float* fcnw = (const float*)d_in[8];
  const float* fcnb = (const float*)d_in[9];
  float* out = (float*)d_out;           // [0,131072): out ; [131072,262144): label

  // ---- workspace carve (256B aligned) ----
  char* p = (char*)d_ws;
  auto carve = [&](size_t bytes) -> char* {
    char* r = p;
    p += (bytes + 255) & ~(size_t)255;
    return r;
  };
  float*          conv1o   = (float*)carve((size_t)NTOT * 6272 * 4);          // 28.9 MB
  __hip_bfloat16* x_bf     = (__hip_bfloat16*)carve((size_t)NTOT * FEAT * 2); // 7.2 MB
  __hip_bfloat16* fc1w_bf  = (__hip_bfloat16*)carve((size_t)LATENT * FEAT * 2);
  __hip_bfloat16* fcnw_bf  = (__hip_bfloat16*)carve((size_t)LABELN * LATENT * 2);
  float*          fc1o     = (float*)carve((size_t)NTOT * LATENT * 4);
  __hip_bfloat16* lat_bf   = (__hip_bfloat16*)carve((size_t)BATCHN * LATENT * 2);
  float*          rep      = (float*)carve((size_t)LABELN * LATENT * 4);
  __hip_bfloat16* rep_bf   = (__hip_bfloat16*)carve((size_t)LABELN * LATENT * 2);
  float*          s_a      = (float*)carve((size_t)BATCHN * LATENT * 4);
  float*          s_b      = (float*)carve((size_t)BATCHN * LATENT * 4);
  __hip_bfloat16* s_a_bf   = (__hip_bfloat16*)carve((size_t)BATCHN * LATENT * 2);
  __hip_bfloat16* s_b_bf   = (__hip_bfloat16*)carve((size_t)BATCHN * LATENT * 2);
  float*          zbuf     = (float*)carve((size_t)BATCHN * LATENT * 4);
  __hip_bfloat16* w_bf     = (__hip_bfloat16*)carve((size_t)LATENT * LATENT * 2);
  float*          min_e    = (float*)carve((size_t)BATCHN * 4);
  __hip_bfloat16* min_s_bf = (__hip_bfloat16*)carve((size_t)BATCHN * LATENT * 2);
  float*          proj     = (float*)carve((size_t)BATCHN * LABELN * 4);
  float*          partial  = (float*)carve(64 * 4);
  float*          rho      = (float*)carve(256);

  auto gemm = [&](float* C, const __hip_bfloat16* A, const __hip_bfloat16* Bt,
                  int M, int N, int K) {
    int waves = (M >> 5) * (N >> 5);     // one wave per 32x32 block
    int blocks = (waves + 7) >> 3;
    gemm_bf16_wmma<<<dim3(blocks), dim3(256), 0, stream>>>(C, A, Bt, M, N, K);
  };

  // ---- weight precision conversion ----
  cvt_bf16<<<dim3((LATENT * FEAT + 255) / 256), dim3(256), 0, stream>>>(fc1w, fc1w_bf, LATENT * FEAT);
  cvt_bf16<<<dim3((LABELN * LATENT + 255) / 256), dim3(256), 0, stream>>>(fcnw, fcnw_bf, LABELN * LATENT);

  // ---- encoder ----
  conv1_relu_pool<<<dim3(NTOT * 32), dim3(256), 0, stream>>>(img, limg, c1w, c1b, conv1o);
  conv2_relu_pool<<<dim3(NTOT * 4), dim3(256), 0, stream>>>(conv1o, c2w, c2b, x_bf);
  gemm(fc1o, x_bf, fc1w_bf, NTOT, LATENT, FEAT);                 // fc1
  fc1_epilogue<<<dim3((NTOT * LATENT + 255) / 256), dim3(256), 0, stream>>>(
      fc1o, fc1b, lat_bf, s_a, s_a_bf, rep, rep_bf);

  // ---- label head: softmax(lat @ fcn_w^T + b) ----
  gemm(proj, lat_bf, fcnw_bf, BATCHN, LABELN, LATENT);
  softmax128<<<dim3(BATCHN), dim3(128), 0, stream>>>(proj, fcnb, 0, out + (size_t)BATCHN * LABELN);

  // ---- Hopfield weight ----
  reduce_part<<<dim3(64), dim3(256), 0, stream>>>(rep, LABELN * LATENT, partial);
  reduce_final<<<dim3(1), dim3(64), 0, stream>>>(partial, 64, 1.0f / (LABELN * LATENT), rho);
  hopfield_w<<<dim3(LATENT * LATENT / 256), dim3(256), 0, stream>>>(rep, rho, w_bf);

  // ---- relaxation: 512 steps; w symmetric, so the energy matmul of step k
  //      is the sign matmul of step k+1 -> one GEMM + one fused kernel/step.
  init_min<<<dim3((BATCHN * LATENT + 255) / 256), dim3(256), 0, stream>>>(min_e, min_s_bf);
  gemm(zbuf, s_a_bf, w_bf, BATCHN, LATENT, LATENT);              // z0 = s0 @ w
  sign_update<<<dim3(BATCHN * LATENT / 256), dim3(256), 0, stream>>>(s_a, zbuf, s_b, s_b_bf); // s1
  float* sc = s_b;            __hip_bfloat16* sc_bf = s_b_bf;    // holds s_k
  float* sn = s_a;            __hip_bfloat16* sn_bf = s_a_bf;
  for (int k = 1; k <= LATENT; ++k) {
    gemm(zbuf, sc_bf, w_bf, BATCHN, LATENT, LATENT);             // z_k = s_k @ w
    energy_sign<<<dim3(BATCHN), dim3(128), 0, stream>>>(zbuf, sc, min_e, min_s_bf, sn, sn_bf);
    float* tf = sc; sc = sn; sn = tf;
    __hip_bfloat16* tb = sc_bf; sc_bf = sn_bf; sn_bf = tb;
  }

  // ---- output head: softmax(|min_s @ rep^T|) ----
  gemm(proj, min_s_bf, rep_bf, BATCHN, LABELN, LATENT);
  softmax128<<<dim3(BATCHN), dim3(128), 0, stream>>>(proj, nullptr, 1, out);
}
